// Trajectory2Seq_20323785244963
// MI455X (gfx1250) — compile-verified, matrix operand order checked by reading, not run on hardware
//
#include <hip/hip_runtime.h>

// ---------------------------------------------------------------------------
// Trajectory2Seq on MI455X (gfx1250): bf16 WMMA GEMMs + fused GRU epilogues.
// N=128, H=512, L=512, T=128, S=64.
// ---------------------------------------------------------------------------

#define N_B   128
#define H_D   512
#define H3_D  1536
#define L_SRC 512
#define T_MAX 128
#define S_SYM 64

typedef unsigned short bfraw;                                      // bf16 bits
typedef __attribute__((ext_vector_type(8)))  unsigned short v8us;
typedef __attribute__((ext_vector_type(16))) unsigned short v16us;
typedef __attribute__((ext_vector_type(16))) __bf16         v16bf;
typedef __attribute__((ext_vector_type(8)))  float          v8f;

// ---- bf16 <-> f32 helpers (bit-level, RNE) --------------------------------
__device__ __forceinline__ bfraw f2br(float f) {
  union { float f; unsigned u; } c; c.f = f;
  unsigned u = c.u;
  return (bfraw)((u + 0x7FFFu + ((u >> 16) & 1u)) >> 16);
}
__device__ __forceinline__ float br2f(bfraw b) {
  union { unsigned u; float f; } c; c.u = ((unsigned)b) << 16; return c.f;
}
__device__ __forceinline__ float brlo2f(unsigned p) {
  union { unsigned u; float f; } c; c.u = p << 16; return c.f;
}
__device__ __forceinline__ float brhi2f(unsigned p) {
  union { unsigned u; float f; } c; c.u = p & 0xFFFF0000u; return c.f;
}
__device__ __forceinline__ float sigmoidf(float x) {
  return 1.0f / (1.0f + __expf(-x));
}

// ---- WMMA fragment load ---------------------------------------------------
// 16-bit A/B lane layout (ISA 7.12.2): lane l holds row (l&15); lanes 0-15
// carry K = {k0..k0+7, k0+16..k0+23}, lanes 16-31 carry K = {k0+8..k0+15,
// k0+24..k0+31}.  Row-major source, 16B-aligned chunks -> global_load_b128.
__device__ __forceinline__ v16bf load_frag(const bfraw* p, int row, int ld, int k0) {
  const int half = (threadIdx.x >> 4) & 1;
  const bfraw* c0 = p + (size_t)row * ld + k0 + half * 8;
  v8us lo = *(const v8us*)c0;
  v8us hi = *(const v8us*)(c0 + 16);
  v16us v = __builtin_shufflevector(lo, hi,
      0, 1, 2, 3, 4, 5, 6, 7, 8, 9, 10, 11, 12, 13, 14, 15);
  return __builtin_bit_cast(v16bf, v);
}

__device__ __forceinline__ v8f wmma_bf16(v16bf a, v16bf b, v8f c) {
  // D = A x B + C, f32 accumulate.  emits v_wmma_f32_16x16x32_bf16
  return __builtin_amdgcn_wmma_f32_16x16x32_bf16(
      false, a, false, b, (short)0, c, false, false);
}

// ---------------------------------------------------------------------------
// prep kernels
// ---------------------------------------------------------------------------
__global__ void cvt_bf16_kernel(const float* __restrict__ s, bfraw* __restrict__ d, int n) {
  int i = blockIdx.x * blockDim.x + threadIdx.x;
  if (i < n) d[i] = f2br(s[i]);
}

// E[t][n][h] = emb[tok(t,n)][h]; tok = SOS(=0) at t=0 else target[n][t-1]
__global__ void embed_gather_kernel(const int* __restrict__ target,
                                    const float* __restrict__ emb,
                                    bfraw* __restrict__ E) {
  int i = blockIdx.x * blockDim.x + threadIdx.x;          // < T*N*H = 2^23
  if (i >= T_MAX * N_B * H_D) return;
  int h = i & (H_D - 1);
  int n = (i >> 9) & (N_B - 1);
  int t = i >> 16;
  int tok = (t == 0) ? 0 : target[n * T_MAX + (t - 1)];
  E[i] = f2br(emb[tok * H_D + h]);
}

__global__ void zero_h_kernel(float* __restrict__ hf, bfraw* __restrict__ hb, int n) {
  int i = blockIdx.x * blockDim.x + threadIdx.x;
  if (i < n) { hf[i] = 0.0f; hb[i] = 0; }
}

__global__ void copy_f32_kernel(const float* __restrict__ s, float* __restrict__ d, int n) {
  int i = blockIdx.x * blockDim.x + threadIdx.x;
  if (i < n) d[i] = s[i];
}

// ---------------------------------------------------------------------------
// Encoder step: gh = h_t @ Whh^T (128x1536, K=512) via WMMA; fused GRU gates.
// 256 wave-tasks = 8 row-tiles x 32 col-tiles; each wave owns r/z/n for its
// 16x16 hidden tile (3 accumulators sharing the A fragment).
// gi (input part, K=2) is folded into the epilogue with scalar FMAs.
// ---------------------------------------------------------------------------
__global__ void __launch_bounds__(256) enc_step_kernel(
    const float* __restrict__ x,        // (N,2,L)
    const float* __restrict__ Wih,      // (3H,2)
    const float* __restrict__ bih,      // (3H)
    const float* __restrict__ bhh,      // (3H)
    const bfraw* __restrict__ Whh_b,    // (3H,H) row-major
    const float* __restrict__ h_in_f,   // (N,H)
    const bfraw* __restrict__ h_in_b,   // (N,H)
    float* __restrict__ h_out_f,
    bfraw* __restrict__ h_out_b,
    bfraw* __restrict__ enc_out_b,      // (N,L,H)
    int t) {
  const int lane = threadIdx.x & 31;
  const int wave = threadIdx.x >> 5;
  const int task = blockIdx.x * 8 + wave;   // 0..255
  const int mi = task >> 5;                 // row tile 0..7
  const int nj = task & 31;                 // col tile 0..31 (hidden dim)
  const int half = lane >> 4;
  const int lr = lane & 15;
  const int mrow = mi * 16 + lr;
  const int jcol = nj * 16 + lr;

  v8f ar = {}, az = {}, an = {};
  for (int k0 = 0; k0 < H_D; k0 += 32) {
    v16bf a  = load_frag(h_in_b, mrow, H_D, k0);
    v16bf br = load_frag(Whh_b, jcol,            H_D, k0);
    v16bf bz = load_frag(Whh_b, jcol + H_D,      H_D, k0);
    v16bf bn = load_frag(Whh_b, jcol + 2 * H_D,  H_D, k0);
    ar = wmma_bf16(a, br, ar);
    az = wmma_bf16(a, bz, az);
    an = wmma_bf16(a, bn, an);
  }

#pragma unroll
  for (int r = 0; r < 8; ++r) {
    const int m = mi * 16 + half * 8 + r;     // sample (C/D layout: M = r+8*half)
    const int j = nj * 16 + lr;               // hidden unit
    const float x0 = x[(m * 2 + 0) * L_SRC + t];
    const float x1 = x[(m * 2 + 1) * L_SRC + t];
    const float gir = x0 * Wih[2 * j]              + x1 * Wih[2 * j + 1]              + bih[j];
    const float giz = x0 * Wih[2 * (j + H_D)]      + x1 * Wih[2 * (j + H_D) + 1]      + bih[j + H_D];
    const float gin = x0 * Wih[2 * (j + 2 * H_D)]  + x1 * Wih[2 * (j + 2 * H_D) + 1]  + bih[j + 2 * H_D];
    const float ghr = ar[r] + bhh[j];
    const float ghz = az[r] + bhh[j + H_D];
    const float ghn = an[r] + bhh[j + 2 * H_D];
    const float rg = sigmoidf(gir + ghr);
    const float zg = sigmoidf(giz + ghz);
    const float nn = tanhf(gin + rg * ghn);
    const float hp = h_in_f[m * H_D + j];
    const float hn = (1.0f - zg) * nn + zg * hp;
    h_out_f[m * H_D + j] = hn;
    const bfraw hb = f2br(hn);
    h_out_b[m * H_D + j] = hb;
    enc_out_b[((size_t)m * L_SRC + t) * H_D + j] = hb;    // (N,L,H)
  }
}

// ---------------------------------------------------------------------------
// Decoder GRU step: gi = e_t @ Wih^T and gh = h_t @ Whh^T fused in one pass;
// 6 f32 accumulators per wave.  q is written to h ping-pong buffers and to
// the left half of the concat([q, attn]) activation matrix.
// ---------------------------------------------------------------------------
__global__ void __launch_bounds__(256) dec_gru_kernel(
    const bfraw* __restrict__ Et,       // (N,H) embeddings for this step
    const bfraw* __restrict__ Wih_b,    // (3H,H)
    const bfraw* __restrict__ Whh_b,    // (3H,H)
    const float* __restrict__ bih,
    const float* __restrict__ bhh,
    const float* __restrict__ h_in_f,
    const bfraw* __restrict__ h_in_b,
    float* __restrict__ h_out_f,
    bfraw* __restrict__ h_out_b,
    bfraw* __restrict__ A_ff) {         // (N,2H), cols [0,H) = q
  const int lane = threadIdx.x & 31;
  const int wave = threadIdx.x >> 5;
  const int task = blockIdx.x * 8 + wave;
  const int mi = task >> 5;
  const int nj = task & 31;
  const int half = lane >> 4;
  const int lr = lane & 15;
  const int mrow = mi * 16 + lr;
  const int jcol = nj * 16 + lr;

  v8f ir = {}, iz = {}, in_ = {}, hr = {}, hz = {}, hn_ = {};
  for (int k0 = 0; k0 < H_D; k0 += 32) {
    v16bf ah = load_frag(h_in_b, mrow, H_D, k0);
    v16bf ae = load_frag(Et,     mrow, H_D, k0);
    v16bf bir = load_frag(Wih_b, jcol,           H_D, k0);
    v16bf biz = load_frag(Wih_b, jcol + H_D,     H_D, k0);
    v16bf bin = load_frag(Wih_b, jcol + 2 * H_D, H_D, k0);
    v16bf bhr = load_frag(Whh_b, jcol,           H_D, k0);
    v16bf bhz = load_frag(Whh_b, jcol + H_D,     H_D, k0);
    v16bf bhn = load_frag(Whh_b, jcol + 2 * H_D, H_D, k0);
    ir  = wmma_bf16(ae, bir, ir);
    iz  = wmma_bf16(ae, biz, iz);
    in_ = wmma_bf16(ae, bin, in_);
    hr  = wmma_bf16(ah, bhr, hr);
    hz  = wmma_bf16(ah, bhz, hz);
    hn_ = wmma_bf16(ah, bhn, hn_);
  }

#pragma unroll
  for (int r = 0; r < 8; ++r) {
    const int m = mi * 16 + half * 8 + r;
    const int j = nj * 16 + lr;
    const float gir = ir[r]  + bih[j];
    const float giz = iz[r]  + bih[j + H_D];
    const float gin = in_[r] + bih[j + 2 * H_D];
    const float ghr = hr[r]  + bhh[j];
    const float ghz = hz[r]  + bhh[j + H_D];
    const float ghn = hn_[r] + bhh[j + 2 * H_D];
    const float rg = sigmoidf(gir + ghr);
    const float zg = sigmoidf(giz + ghz);
    const float nn = tanhf(gin + rg * ghn);
    const float hp = h_in_f[m * H_D + j];
    const float q  = (1.0f - zg) * nn + zg * hp;
    h_out_f[m * H_D + j] = q;
    const bfraw qb = f2br(q);
    h_out_b[m * H_D + j] = qb;
    A_ff[m * (2 * H_D) + j] = qb;
  }
}

// ---------------------------------------------------------------------------
// Attention: per-sample GEMV scores (L x H) * q, softmax over L (LDS
// reductions), context = w^T * enc_out.  One block per sample.  enc_out is
// bf16 and L2-resident (64 MB < 192 MB global L2).
// ---------------------------------------------------------------------------
__global__ void __launch_bounds__(256) attn_kernel(
    const bfraw* __restrict__ q_b,      // (N,H)
    const bfraw* __restrict__ enc_b,    // (N,L,H)
    float* __restrict__ all_w,          // (N,T,L) output
    bfraw* __restrict__ A_ff,           // (N,2H), cols [H,2H) = context
    int t) {
  __shared__ float qs[H_D];
  __shared__ float sc[L_SRC];
  __shared__ float red[16];
  const int n = blockIdx.x;
  const int tid = threadIdx.x;
  const int lane = tid & 31;
  const int wv = tid >> 5;

  for (int i = tid; i < H_D; i += 256) qs[i] = br2f(q_b[n * H_D + i]);
  __syncthreads();

  const bfraw* e = enc_b + (size_t)n * L_SRC * H_D;
  for (int l = tid; l < L_SRC; l += 256) {
    const unsigned* rw = (const unsigned*)(e + (size_t)l * H_D);  // 2 bf16/word
    float s = 0.0f;
#pragma unroll 4
    for (int hh = 0; hh < H_D / 2; ++hh) {
      unsigned p = rw[hh];
      s += brlo2f(p) * qs[2 * hh] + brhi2f(p) * qs[2 * hh + 1];
    }
    sc[l] = s;
  }
  __syncthreads();

  // softmax: block max
  float mx = -3.4e38f;
  for (int l = tid; l < L_SRC; l += 256) mx = fmaxf(mx, sc[l]);
  for (int o = 16; o > 0; o >>= 1) mx = fmaxf(mx, __shfl_xor(mx, o, 32));
  if (lane == 0) red[wv] = mx;
  __syncthreads();
  float mxall = red[0];
#pragma unroll
  for (int i = 1; i < 8; ++i) mxall = fmaxf(mxall, red[i]);

  float sm = 0.0f;
  for (int l = tid; l < L_SRC; l += 256) {
    float ev = __expf(sc[l] - mxall);
    sc[l] = ev;
    sm += ev;
  }
  for (int o = 16; o > 0; o >>= 1) sm += __shfl_xor(sm, o, 32);
  if (lane == 0) red[8 + wv] = sm;
  __syncthreads();
  float tot = 0.0f;
#pragma unroll
  for (int i = 0; i < 8; ++i) tot += red[8 + i];
  const float inv = 1.0f / tot;

  for (int l = tid; l < L_SRC; l += 256) {
    float w = sc[l] * inv;
    sc[l] = w;
    all_w[((size_t)n * T_MAX + t) * L_SRC + l] = w;
  }
  __syncthreads();

  // context: attn[h] = sum_l w[l] * enc[l][h]  (threads sweep h, coalesced)
  for (int h = tid; h < H_D; h += 256) {
    float acc = 0.0f;
    for (int l = 0; l < L_SRC; ++l) acc += sc[l] * br2f(e[(size_t)l * H_D + h]);
    A_ff[n * (2 * H_D) + H_D + h] = f2br(acc);
  }
}

// ---------------------------------------------------------------------------
// FF (K=2H) + output head (K=H) fused through LDS.  One block per 16 samples:
// phase 1: 8 waves x 4 tiles -> ff (16x512) staged bf16 in LDS (ds path);
// phase 2: waves 0-3 each produce one 16x16 tile of dec (S=64) via WMMA with
// the A fragment sourced from LDS (ds_load_b128).
// ---------------------------------------------------------------------------
__global__ void __launch_bounds__(256) ff_fc_kernel(
    const bfraw* __restrict__ A_ff,     // (N,2H)
    const bfraw* __restrict__ Wff_b,    // (H,2H)
    const float* __restrict__ bff,      // (H)
    const bfraw* __restrict__ Wfc_b,    // (S,H)
    const float* __restrict__ bfc,      // (S)
    float* __restrict__ v_out,          // (N,T,S)
    int t) {
  __shared__ bfraw ffs[16 * H_D];       // 16 KB
  const int lane = threadIdx.x & 31;
  const int wave = threadIdx.x >> 5;
  const int half = lane >> 4;
  const int lr = lane & 15;
  const int mb = blockIdx.x;            // 0..7 -> 16 samples each

  // phase 1: ff = [q|attn] @ Wff^T + bff
#pragma unroll
  for (int c = 0; c < 4; ++c) {
    const int ct = wave * 4 + c;        // col tile 0..31
    const int mrow = mb * 16 + lr;
    const int jcol = ct * 16 + lr;
    v8f acc = {};
    for (int k0 = 0; k0 < 2 * H_D; k0 += 32) {
      v16bf a = load_frag(A_ff,  mrow, 2 * H_D, k0);
      v16bf b = load_frag(Wff_b, jcol, 2 * H_D, k0);
      acc = wmma_bf16(a, b, acc);
    }
#pragma unroll
    for (int r = 0; r < 8; ++r) {
      const int ml = half * 8 + r;
      const int j = ct * 16 + lr;
      ffs[ml * H_D + j] = f2br(acc[r] + bff[j]);
    }
  }
  __syncthreads();

  // phase 2: dec = ff @ Wfc^T + bfc  (A from LDS)
  if (wave < 4) {
    const int srow = lr;                // local sample row in ffs
    const int scol = wave * 16 + lr;    // symbol row of Wfc
    v8f acc = {};
    for (int k0 = 0; k0 < H_D; k0 += 32) {
      v16bf a = load_frag(ffs,   srow, H_D, k0);   // ds_load_b128
      v16bf b = load_frag(Wfc_b, scol, H_D, k0);
      acc = wmma_bf16(a, b, acc);
    }
#pragma unroll
    for (int r = 0; r < 8; ++r) {
      const int m = mb * 16 + half * 8 + r;
      const int s = wave * 16 + lr;
      v_out[((size_t)m * T_MAX + t) * S_SYM + s] = acc[r] + bfc[s];
    }
  }
}

// ---------------------------------------------------------------------------
// host side
// ---------------------------------------------------------------------------
extern "C" void kernel_launch(void* const* d_in, const int* in_sizes, int n_in,
                              void* d_out, int out_size, void* d_ws, size_t ws_size,
                              hipStream_t stream) {
  (void)in_sizes; (void)n_in; (void)out_size; (void)ws_size;
  const float* x        = (const float*)d_in[0];
  const int*   target   = (const int*)d_in[1];
  const float* enc_Wih  = (const float*)d_in[2];
  const float* enc_Whh  = (const float*)d_in[3];
  const float* enc_bih  = (const float*)d_in[4];
  const float* enc_bhh  = (const float*)d_in[5];
  const float* dec_Wih  = (const float*)d_in[6];
  const float* dec_Whh  = (const float*)d_in[7];
  const float* dec_bih  = (const float*)d_in[8];
  const float* dec_bhh  = (const float*)d_in[9];
  const float* emb      = (const float*)d_in[10];
  const float* attn_W   = (const float*)d_in[11];
  const float* attn_b   = (const float*)d_in[12];
  const float* fc_W     = (const float*)d_in[13];
  const float* fc_b     = (const float*)d_in[14];
  float* out = (float*)d_out;

  // output layout: v_out (N*T*S) | dec_hN (N*H) | all_w (N*T*L)
  float* v_out  = out;
  float* dec_hN = out + (size_t)N_B * T_MAX * S_SYM;
  float* all_w  = dec_hN + (size_t)N_B * H_D;

  // workspace carve-out (256B aligned)
  char* wp = (char*)d_ws;
  auto carve = [&](size_t bytes) -> void* {
    void* p = (void*)wp;
    wp += (bytes + 255) & ~(size_t)255;
    return p;
  };
  bfraw* Whh_enc_b = (bfraw*)carve((size_t)H3_D * H_D * 2);
  bfraw* Wih_dec_b = (bfraw*)carve((size_t)H3_D * H_D * 2);
  bfraw* Whh_dec_b = (bfraw*)carve((size_t)H3_D * H_D * 2);
  bfraw* Wff_b     = (bfraw*)carve((size_t)H_D * 2 * H_D * 2);
  bfraw* Wfc_b     = (bfraw*)carve((size_t)S_SYM * H_D * 2);
  bfraw* E_b       = (bfraw*)carve((size_t)T_MAX * N_B * H_D * 2);
  bfraw* enc_out_b = (bfraw*)carve((size_t)N_B * L_SRC * H_D * 2);
  float* hf[2];
  bfraw* hb[2];
  hf[0] = (float*)carve((size_t)N_B * H_D * 4);
  hf[1] = (float*)carve((size_t)N_B * H_D * 4);
  hb[0] = (bfraw*)carve((size_t)N_B * H_D * 2);
  hb[1] = (bfraw*)carve((size_t)N_B * H_D * 2);
  bfraw* A_ff = (bfraw*)carve((size_t)N_B * 2 * H_D * 2);

  // --- prep: weight conversions, embedding gather, h0 = 0 ---
  cvt_bf16_kernel<<<(H3_D * H_D + 255) / 256, 256, 0, stream>>>(enc_Whh, Whh_enc_b, H3_D * H_D);
  cvt_bf16_kernel<<<(H3_D * H_D + 255) / 256, 256, 0, stream>>>(dec_Wih, Wih_dec_b, H3_D * H_D);
  cvt_bf16_kernel<<<(H3_D * H_D + 255) / 256, 256, 0, stream>>>(dec_Whh, Whh_dec_b, H3_D * H_D);
  cvt_bf16_kernel<<<(H_D * 2 * H_D + 255) / 256, 256, 0, stream>>>(attn_W, Wff_b, H_D * 2 * H_D);
  cvt_bf16_kernel<<<(S_SYM * H_D + 255) / 256, 256, 0, stream>>>(fc_W, Wfc_b, S_SYM * H_D);
  embed_gather_kernel<<<(T_MAX * N_B * H_D + 255) / 256, 256, 0, stream>>>(target, emb, E_b);
  zero_h_kernel<<<(N_B * H_D + 255) / 256, 256, 0, stream>>>(hf[0], hb[0], N_B * H_D);

  // --- encoder scan: 512 sequential WMMA steps (ping-pong h) ---
  for (int t = 0; t < L_SRC; ++t) {
    enc_step_kernel<<<32, 256, 0, stream>>>(
        x, enc_Wih, enc_bih, enc_bhh, Whh_enc_b,
        hf[t & 1], hb[t & 1], hf[(t + 1) & 1], hb[(t + 1) & 1],
        enc_out_b, t);
  }
  // after 512 steps the final hidden state sits in buffer 0

  // --- decoder scan: GRU -> attention -> FF+head, 128 steps ---
  for (int t = 0; t < T_MAX; ++t) {
    dec_gru_kernel<<<32, 256, 0, stream>>>(
        E_b + (size_t)t * N_B * H_D, Wih_dec_b, Whh_dec_b,
        dec_bih, dec_bhh,
        hf[t & 1], hb[t & 1], hf[(t + 1) & 1], hb[(t + 1) & 1],
        A_ff);
    attn_kernel<<<N_B, 256, 0, stream>>>(
        hb[(t + 1) & 1], enc_out_b, all_w, A_ff, t);
    ff_fc_kernel<<<8, 256, 0, stream>>>(
        A_ff, Wff_b, attn_b, Wfc_b, fc_b, v_out, t);
  }

  // final decoder hidden state (t=127 wrote buffer 0)
  copy_f32_kernel<<<(N_B * H_D + 255) / 256, 256, 0, stream>>>(
      hf[0], dec_hN, N_B * H_D);
}
